// GraphCastEncoder_77532749627487
// MI455X (gfx1250) — matrix operand order repaired and checked by reading, not compiled, not run on hardware
//
#include <hip/hip_runtime.h>
#include <stdint.h>

// ---------------------------------------------------------------------------
// CDNA5 (gfx1250) GraphCast encoder: bf16 WMMA (f32 accumulate) MLPs + gather.
// ---------------------------------------------------------------------------

typedef __attribute__((ext_vector_type(16))) __bf16   v16bf;
typedef __attribute__((ext_vector_type(8)))  float    v8f;
typedef __attribute__((ext_vector_type(4)))  uint32_t u32x4;

union BF16Frag {
  uint32_t u[8];
  u32x4    q[2];
  v16bf    v;
};

__device__ __forceinline__ uint16_t f2bf(float f) {
  uint32_t u = __float_as_uint(f);
  u += 0x7FFFu + ((u >> 16) & 1u);           // round-to-nearest-even
  return (uint16_t)(u >> 16);
}
__device__ __forceinline__ float bf2f(uint16_t h) {
  return __uint_as_float(((uint32_t)h) << 16);
}

#define LDIM    512      // hidden / output width
#define NTILES  32       // 512 / 16 N-tiles
#define WAVES   8
#define NTW     4        // N-tiles per wave
#define MTILES  2        // M-tiles per block
#define ROWS    32       // rows per block (MTILES*16)
#define SB      514      // LDS stride (bf16 elems) for stage-B activations

// ---------------------------------------------------------------------------
// Pack f32 weights (K x N row-major) into bf16 WMMA B-fragments.
// Fragment order: dword[( (kt*NT + nt)*32 + lane )*8 + r]
// B layout (32x16, 16-bit): dword r packs (K = (r>=4?16:0) + half*8 + (r&3)*2, K+1)
// for column n = nt*16 + (lane&15).
// ---------------------------------------------------------------------------
__global__ void pack_w_kernel(const float* __restrict__ W, uint32_t* __restrict__ out,
                              int K, int Kpad, int N, int total) {
  int t = blockIdx.x * 256 + threadIdx.x;
  if (t >= total) return;
  const int NT   = N >> 4;
  const int lane = t & 31;
  const int tile = t >> 5;
  const int nt   = tile % NT;
  const int kt   = tile / NT;
  const int half = lane >> 4;
  const int n    = nt * 16 + (lane & 15);
  uint32_t* o = out + (size_t)t * 8;
#pragma unroll
  for (int r = 0; r < 8; ++r) {
    int k = kt * 32 + ((r >= 4) ? 16 : 0) + half * 8 + (r & 3) * 2;
    uint32_t lo = (k     < K) ? (uint32_t)f2bf(W[(size_t)k * N + n])       : 0u;
    uint32_t hi = (k + 1 < K) ? (uint32_t)f2bf(W[(size_t)(k + 1) * N + n]) : 0u;
    o[r] = lo | (hi << 16);
  }
  (void)Kpad;
}

// ---------------------------------------------------------------------------
// Fused MLP: out = SiLU(LN(x @ W1 + b1)) @ W2 + b2      (N = hidden = 512)
// 256 threads (8 waves), 32 rows per block.
// ---------------------------------------------------------------------------
template <int K1, int K1PAD, bool IN_BF16, bool OUT_BF16>
__global__ __launch_bounds__(256, 1)
void mlp_wmma_kernel(const void* __restrict__ xin,
                     const uint32_t* __restrict__ w1p, const float* __restrict__ b1,
                     const float* __restrict__ gamma, const float* __restrict__ beta,
                     const uint32_t* __restrict__ w2p, const float* __restrict__ b2,
                     void* __restrict__ out, int M_total) {
  constexpr int SA  = K1PAD + 2;       // stage-A LDS stride (bank-conflict pad)
  constexpr int KT1 = K1PAD / 32;

  __shared__ uint16_t xb[ROWS * 546];  // bf16 activations (A-matrix), both stages
  __shared__ float    psum[WAVES][ROWS];
  __shared__ float    psq[WAVES][ROWS];
  __shared__ float    smu[ROWS];
  __shared__ float    srstd[ROWS];

  const int tid  = threadIdx.x;
  const int wave = tid >> 5;
  const int lane = tid & 31;
  const int half = lane >> 4;
  const int ml   = lane & 15;
  const int row0 = blockIdx.x * ROWS;

  // ---- load input tile into LDS as bf16 --------------------------------
  if (!IN_BF16) {
    const float* xf = (const float*)xin;
    for (int i = tid; i < ROWS * K1; i += 256) {
      int m = i / K1, k = i - m * K1;
      int rg = row0 + m;
      float v = (rg < M_total) ? xf[(size_t)rg * K1 + k] : 0.f;
      xb[m * SA + k] = f2bf(v);
    }
  } else {
    const uint16_t* xh = (const uint16_t*)xin;
    for (int i = tid; i < ROWS * K1PAD; i += 256) {
      int m = i / K1PAD, k = i - m * K1PAD;
      int rg = row0 + m;
      xb[m * SA + k] = (rg < M_total) ? xh[(size_t)rg * K1PAD + k] : (uint16_t)0;
    }
  }
  __syncthreads();

  v8f acc[MTILES][NTW];
#pragma unroll
  for (int mt = 0; mt < MTILES; ++mt)
#pragma unroll
    for (int t = 0; t < NTW; ++t)
#pragma unroll
      for (int r = 0; r < 8; ++r) acc[mt][t][r] = 0.f;

  // ---- GEMM 1: h = x @ W1 ----------------------------------------------
  for (int kt = 0; kt < KT1; ++kt) {
    const int kb = kt * 32;
    BF16Frag a[MTILES];
#pragma unroll
    for (int mt = 0; mt < MTILES; ++mt) {
      const uint16_t* rp = &xb[(mt * 16 + ml) * SA];
#pragma unroll
      for (int r = 0; r < 8; ++r) {
        // A layout (16x32, 16-bit): K = (r>=4?16:0) + half*8 + (r&3)*2
        int k = kb + ((r >= 4) ? 16 : 0) + half * 8 + (r & 3) * 2;
        a[mt].u[r] = *(const uint32_t*)&rp[k];
      }
    }
#pragma unroll
    for (int t = 0; t < NTW; ++t) {
      const int ntg = wave * NTW + t;
      const u32x4* wp =
          (const u32x4*)&w1p[(((size_t)kt * NTILES + ntg) * 32 + lane) * 8];
      BF16Frag bfr;
      bfr.q[0] = wp[0];
      bfr.q[1] = wp[1];
#pragma unroll
      for (int mt = 0; mt < MTILES; ++mt)
        acc[mt][t] = __builtin_amdgcn_wmma_f32_16x16x32_bf16(
            false, a[mt].v, false, bfr.v, (short)0, acc[mt][t], false, false);
    }
  }

  // ---- + b1, then LayerNorm stats (register-resident) -------------------
#pragma unroll
  for (int t = 0; t < NTW; ++t) {
    const float bias = b1[(wave * NTW + t) * 16 + ml];
#pragma unroll
    for (int mt = 0; mt < MTILES; ++mt)
#pragma unroll
      for (int r = 0; r < 8; ++r) acc[mt][t][r] += bias;
  }
#pragma unroll
  for (int mt = 0; mt < MTILES; ++mt)
#pragma unroll
    for (int r = 0; r < 8; ++r) {
      float s = 0.f, s2 = 0.f;
#pragma unroll
      for (int t = 0; t < NTW; ++t) {
        float v = acc[mt][t][r];
        s += v;
        s2 += v * v;
      }
#pragma unroll
      for (int off = 8; off > 0; off >>= 1) {   // reduce within 16-lane half
        s  += __shfl_xor(s, off, 32);
        s2 += __shfl_xor(s2, off, 32);
      }
      const int m = mt * 16 + half * 8 + r;
      if (ml == 0) { psum[wave][m] = s; psq[wave][m] = s2; }
    }
  __syncthreads();
  if (tid < ROWS) {
    float s = 0.f, s2 = 0.f;
#pragma unroll
    for (int w = 0; w < WAVES; ++w) { s += psum[w][tid]; s2 += psq[w][tid]; }
    const float mu = s * (1.f / 512.f);
    smu[tid]   = mu;
    srstd[tid] = rsqrtf(s2 * (1.f / 512.f) - mu * mu + 1e-5f);
  }
  __syncthreads();

  // ---- normalize + SiLU in registers -> bf16 A-matrix in LDS ------------
#pragma unroll
  for (int t = 0; t < NTW; ++t) {
    const int n = (wave * NTW + t) * 16 + ml;
    const float g = gamma[n], be = beta[n];
#pragma unroll
    for (int mt = 0; mt < MTILES; ++mt)
#pragma unroll
      for (int r = 0; r < 8; ++r) {
        const int m = mt * 16 + half * 8 + r;
        float v = (acc[mt][t][r] - smu[m]) * srstd[m] * g + be;
        v = v / (1.f + __expf(-v));           // SiLU
        xb[m * SB + n] = f2bf(v);
      }
  }
  __syncthreads();

  // ---- GEMM 2: out = h @ W2 --------------------------------------------
#pragma unroll
  for (int mt = 0; mt < MTILES; ++mt)
#pragma unroll
    for (int t = 0; t < NTW; ++t)
#pragma unroll
      for (int r = 0; r < 8; ++r) acc[mt][t][r] = 0.f;

  for (int kt = 0; kt < LDIM / 32; ++kt) {
    const int kb = kt * 32;
    BF16Frag a[MTILES];
#pragma unroll
    for (int mt = 0; mt < MTILES; ++mt) {
      const uint16_t* rp = &xb[(mt * 16 + ml) * SB];
#pragma unroll
      for (int r = 0; r < 8; ++r) {
        int k = kb + ((r >= 4) ? 16 : 0) + half * 8 + (r & 3) * 2;
        a[mt].u[r] = *(const uint32_t*)&rp[k];
      }
    }
#pragma unroll
    for (int t = 0; t < NTW; ++t) {
      const int ntg = wave * NTW + t;
      const u32x4* wp =
          (const u32x4*)&w2p[(((size_t)kt * NTILES + ntg) * 32 + lane) * 8];
      BF16Frag bfr;
      bfr.q[0] = wp[0];
      bfr.q[1] = wp[1];
#pragma unroll
      for (int mt = 0; mt < MTILES; ++mt)
        acc[mt][t] = __builtin_amdgcn_wmma_f32_16x16x32_bf16(
            false, a[mt].v, false, bfr.v, (short)0, acc[mt][t], false, false);
    }
  }

  // ---- + b2, store -------------------------------------------------------
#pragma unroll
  for (int t = 0; t < NTW; ++t) {
    const int n = (wave * NTW + t) * 16 + ml;
    const float bias = b2[n];
#pragma unroll
    for (int mt = 0; mt < MTILES; ++mt)
#pragma unroll
      for (int r = 0; r < 8; ++r) {
        const int rg = row0 + mt * 16 + half * 8 + r;
        if (rg < M_total) {
          const float v = acc[mt][t][r] + bias;
          if (OUT_BF16)
            ((uint16_t*)out)[(size_t)rg * LDIM + n] = f2bf(v);
          else
            ((float*)out)[(size_t)rg * LDIM + n] = v;
        }
      }
  }
}

// ---------------------------------------------------------------------------
// Gather + weighted pool + concat(mesh features), bf16 out (544-col padded).
// One block per (b, m) mesh node.
// ---------------------------------------------------------------------------
__global__ __launch_bounds__(256)
void gather_pool_kernel(const uint16_t* __restrict__ gp,
                        const int* __restrict__ idx, const float* __restrict__ w,
                        const float* __restrict__ meshf,
                        uint16_t* __restrict__ combined, int NM, int NG) {
  const int bm = blockIdx.x;
  const int b  = bm / NM;
  const int m  = bm - b * NM;
  __shared__ int   sidx[16];
  __shared__ float sw[16];
  const int tid = threadIdx.x;
  if (tid < 16) {
    sidx[tid] = idx[m * 16 + tid];
    sw[tid]   = w[m * 16 + tid];
  }
  __syncthreads();
  const size_t base_b = (size_t)b * NG;
  const size_t orow   = (size_t)bm * 544;
  for (int l = tid; l < 512; l += 256) {
    float acc = 0.f;
#pragma unroll
    for (int k = 0; k < 16; ++k)
      acc += sw[k] * bf2f(gp[(base_b + (size_t)sidx[k]) * 512 + l]);
    combined[orow + l] = f2bf(acc);
  }
  if (tid < 32) {
    const int c = 512 + tid;
    const float v = (c < 515) ? meshf[(size_t)m * 3 + (c - 512)] : 0.f;
    combined[orow + c] = f2bf(v);
  }
}

// ---------------------------------------------------------------------------
extern "C" void kernel_launch(void* const* d_in, const int* in_sizes, int n_in,
                              void* d_out, int out_size, void* d_ws, size_t ws_size,
                              hipStream_t stream) {
  (void)n_in; (void)out_size; (void)ws_size;

  const float* grid_data = (const float*)d_in[0];
  const float* meshf     = (const float*)d_in[1];
  const int*   g2m_idx   = (const int*)d_in[2];
  const float* g2m_w     = (const float*)d_in[3];
  const float* gW1 = (const float*)d_in[4];
  const float* gb1 = (const float*)d_in[5];
  const float* gga = (const float*)d_in[6];
  const float* gbe = (const float*)d_in[7];
  const float* gW2 = (const float*)d_in[8];
  const float* gb2 = (const float*)d_in[9];
  const float* cW1 = (const float*)d_in[10];
  const float* cb1 = (const float*)d_in[11];
  const float* cga = (const float*)d_in[12];
  const float* cbe = (const float*)d_in[13];
  const float* cW2 = (const float*)d_in[14];
  const float* cb2 = (const float*)d_in[15];

  const int B = 2, NG = 65536, GD = 256;
  const int NM = in_sizes[2] / 16;   // 10242
  const int Mg = B * NG;             // 131072 grid rows
  const int Mc = B * NM;             // 20484 mesh rows
  const int KCPAD = 544;             // 515 padded to 17 K-tiles of 32

  // workspace layout (all 256B aligned)
  char* ws = (char*)d_ws;
  uint32_t* gW1p = (uint32_t*)ws;                 ws += (size_t)(GD  / 32) * 32 * 32 * 8 * 4;  // 256 KB
  uint32_t* gW2p = (uint32_t*)ws;                 ws += (size_t)(512 / 32) * 32 * 32 * 8 * 4;  // 512 KB
  uint32_t* cW1p = (uint32_t*)ws;                 ws += (size_t)(KCPAD / 32) * 32 * 32 * 8 * 4;// 544 KB
  uint32_t* cW2p = (uint32_t*)ws;                 ws += (size_t)(512 / 32) * 32 * 32 * 8 * 4;  // 512 KB
  uint16_t* gp   = (uint16_t*)ws;                 ws += (size_t)Mg * 512 * 2;                  // 134 MB
  uint16_t* comb = (uint16_t*)ws;                 // Mc * 544 * 2 ~ 22 MB

  // 1) pack all weights to bf16 fragment layout
  {
    int t1 = (GD / 32) * 32 * 32;
    pack_w_kernel<<<(t1 + 255) / 256, 256, 0, stream>>>(gW1, gW1p, GD, GD, 512, t1);
    int t2 = (512 / 32) * 32 * 32;
    pack_w_kernel<<<(t2 + 255) / 256, 256, 0, stream>>>(gW2, gW2p, 512, 512, 512, t2);
    int t3 = (KCPAD / 32) * 32 * 32;
    pack_w_kernel<<<(t3 + 255) / 256, 256, 0, stream>>>(cW1, cW1p, 515, KCPAD, 512, t3);
    pack_w_kernel<<<(t2 + 255) / 256, 256, 0, stream>>>(cW2, cW2p, 512, 512, 512, t2);
  }

  // 2) grid MLP: (Mg x 256) -> (Mg x 512) bf16
  mlp_wmma_kernel<256, 256, false, true>
      <<<Mg / ROWS, 256, 0, stream>>>(grid_data, gW1p, gb1, gga, gbe, gW2p, gb2, gp, Mg);

  // 3) gather + pool + concat -> combined (Mc x 544) bf16
  gather_pool_kernel<<<Mc, 256, 0, stream>>>(gp, g2m_idx, g2m_w, meshf, comb, NM, NG);

  // 4) combined MLP: (Mc x 544) -> (Mc x 512) f32 out
  mlp_wmma_kernel<544, 544, true, false>
      <<<(Mc + ROWS - 1) / ROWS, 256, 0, stream>>>(comb, cW1p, cb1, cga, cbe, cW2p, cb2,
                                                   (float*)d_out, Mc);
}
